// GraphEncoder_23965917511889
// MI455X (gfx1250) — matrix operand order repaired and checked by reading, not compiled, not run on hardware
//
#include <hip/hip_runtime.h>
#include <hip/hip_bf16.h>

// ---------------------------------------------------------------------------
// PE-aware GAT encoder for MI455X (gfx1250, wave32, WMMA).
// Only branches reaching the outputs (hm, pe_n) are computed.
//
//  * Dense x@W via v_wmma_f32_16x16x32_f16 on PRE-SWIZZLED fragment-layout
//    operands: inner loop = global_load_b128 clauses + v_wmma, no LDS.
//    2x-unrolled ping-pong register sets -> zero rotation copies, partial
//    loadcnt waits (WMMA overlaps the other set's loads).
//  * Segment softmax/aggregation as a GATHER over a dst-CSR (built once):
//    zero float atomics; coalesced 1KB xp[src] rows (51MB << 192MB L2).
// ---------------------------------------------------------------------------

typedef _Float16 half_t;
typedef __attribute__((ext_vector_type(16))) _Float16 v16h;
typedef __attribute__((ext_vector_type(8)))  float    v8f;

#define GAT_H 4
#define GAT_EMB 256
#define GAT_PED 32
#define NEG_SLOPE 0.2f

#define WMMA_F16(A, B, CC) \
    __builtin_amdgcn_wmma_f32_16x16x32_f16(false, (A), false, (B), (short)0, (CC), false, false)

// ------------------------------ fill kernel --------------------------------
__global__ void fill_i32(int* __restrict__ p, int v, int n) {
    int i = blockIdx.x * blockDim.x + threadIdx.x;
    if (i < n) p[i] = v;
}

// ------------------------- CSR build (dst-sorted) --------------------------
__global__ void deg_kernel(const int* __restrict__ dst, int* __restrict__ deg, int Ee) {
    int e = blockIdx.x * blockDim.x + threadIdx.x;
    if (e < Ee) atomicAdd(&deg[dst[e]], 1);
}

// single-workgroup Hillis-Steele scan (N=50000 -> ~49 chunks; trivial cost)
__global__ __launch_bounds__(1024)
void scan_kernel(const int* __restrict__ deg, int* __restrict__ start, int n) {
    __shared__ int buf[1024];
    __shared__ int carry;
    const int t = threadIdx.x;
    if (t == 0) carry = 0;
    __syncthreads();
    for (int base = 0; base < n; base += 1024) {
        int v = (base + t < n) ? deg[base + t] : 0;
        buf[t] = v;
        __syncthreads();
        for (int off = 1; off < 1024; off <<= 1) {
            int x = (t >= off) ? buf[t - off] : 0;
            __syncthreads();
            buf[t] += x;
            __syncthreads();
        }
        if (base + t < n) start[base + t] = carry + buf[t] - v;  // exclusive
        int total = buf[1023];
        __syncthreads();
        if (t == 0) carry += total;
        __syncthreads();
    }
    if (t == 0) start[n] = carry;
}

__global__ void scatter_kernel(const int* __restrict__ dst, const int* __restrict__ start,
                               int* __restrict__ cursor, int* __restrict__ elist, int Ee) {
    int e = blockIdx.x * blockDim.x + threadIdx.x;
    if (e >= Ee) return;
    int d = dst[e];
    int pos = atomicAdd(&cursor[d], 1);
    elist[start[d] + pos] = e;
}

// ----------------- pre-swizzle A [MxK] f32 -> WMMA A-frag f16 --------------
// CDNA5 16-bit A 16x32: lanes0-15 row=l, VGPR0-3 K=0..7, VGPR4-7 K=16..23;
// lanes16-31 K=8..15 / 24..31. Stored as [rowTile][chunk][lane][16 halfs].
__global__ void swizzleA_kernel(const float* __restrict__ A, half_t* __restrict__ out,
                                int M, int K, int relu) {
    int idx = blockIdx.x * 256 + threadIdx.x;
    if (idx >= M * K) return;
    int r = idx / K, k = idx - r * K;
    float v = A[idx];
    if (relu) v = fmaxf(v, 0.f);
    int rt = r >> 4, l = r & 15;
    int chunk = k >> 5, kk = k & 31;
    int lane = l + (((kk >> 3) & 1) << 4);
    int e    = (kk & 7) + ((kk >> 4) << 3);
    size_t dstIdx = (((size_t)rt * (K >> 5) + chunk) * 32 + lane) * 16 + e;
    out[dstIdx] = (half_t)v;
}

// ----------------- pre-swizzle B [Kx256] f32 -> WMMA B-frag f16 ------------
// CDNA5 16-bit B 32x16: lanes0-15 hold K=0..7,16..23 (col=l), lanes16-31 K=8..15,24..31.
// Stored as [chunk][colTile][lane][16 halfs].
__global__ void swizzleB_kernel(const float* __restrict__ B, half_t* __restrict__ out, int K) {
    int idx = blockIdx.x * 256 + threadIdx.x;
    if (idx >= K * GAT_EMB) return;
    int k = idx >> 8, n = idx & 255;
    float v = B[idx];
    int c = k >> 5, kk = k & 31;
    int ct = n >> 4, l = n & 15;
    int hf = (kk >> 3) & 1, i = (kk & 7) + ((kk >> 4) << 3);
    size_t dstIdx = (((size_t)c * 16 + ct) * 32 + hf * 16 + l) * 16 + i;
    out[dstIdx] = (half_t)v;
}

// ------------- WMMA GEMM on pre-swizzled fragments (LDS-free) --------------
// Block = 32 rows x 256 cols, 8 waves: wave w -> row tile (w&1), col tiles
// (w>>1)*4..+3. 2x-unrolled ping-pong: load set1 | wmma set0 | load set0' |
// wmma set1 -- no register rotation, loads overlap the other set's WMMAs.
__global__ __launch_bounds__(256)
void gemm_wmma_swz(const half_t* __restrict__ Aswz, const half_t* __restrict__ Bswz,
                   float* __restrict__ C, int M, int K) {
    const int tid  = threadIdx.x;
    const int wave = tid >> 5;
    const int lane = tid & 31;
    const int nCh  = K >> 5;                     // even for K=128/256
    const int rtile = blockIdx.x * 2 + (wave & 1);
    const int cgrp  = wave >> 1;                 // 4 col-tiles of 16 each
    if (rtile * 16 >= M) return;

    v8f acc0 = {}, acc1 = {}, acc2 = {}, acc3 = {};
    const v16h* Af = (const v16h*)Aswz + (size_t)rtile * nCh * 32 + lane;
    const v16h* Bf = (const v16h*)Bswz + (size_t)cgrp * 4 * 32 + lane;

    // set0 <- chunk 0
    v16h a0  = Af[0];
    v16h b00 = Bf[0], b01 = Bf[32], b02 = Bf[64], b03 = Bf[96];

    int c = 0;
    for (; c + 1 < nCh; c += 2) {
        // set1 <- chunk c+1 (issued before consuming set0)
        v16h a1 = Af[(size_t)(c + 1) * 32];
        const v16h* bp1 = Bf + (size_t)(c + 1) * 512;
        v16h b10 = bp1[0], b11 = bp1[32], b12 = bp1[64], b13 = bp1[96];

        acc0 = WMMA_F16(a0, b00, acc0);
        acc1 = WMMA_F16(a0, b01, acc1);
        acc2 = WMMA_F16(a0, b02, acc2);
        acc3 = WMMA_F16(a0, b03, acc3);

        if (c + 2 < nCh) {
            // set0 <- chunk c+2 (issued before consuming set1)
            a0 = Af[(size_t)(c + 2) * 32];
            const v16h* bp0 = Bf + (size_t)(c + 2) * 512;
            b00 = bp0[0]; b01 = bp0[32]; b02 = bp0[64]; b03 = bp0[96];
            __builtin_prefetch((const void*)(Af + (size_t)(c + 4) * 32), 0, 1);
        }

        acc0 = WMMA_F16(a1, b10, acc0);
        acc1 = WMMA_F16(a1, b11, acc1);
        acc2 = WMMA_F16(a1, b12, acc2);
        acc3 = WMMA_F16(a1, b13, acc3);
    }
    if (c < nCh) {  // odd-chunk epilogue (not hit for K=128/256)
        acc0 = WMMA_F16(a0, b00, acc0);
        acc1 = WMMA_F16(a0, b01, acc1);
        acc2 = WMMA_F16(a0, b02, acc2);
        acc3 = WMMA_F16(a0, b03, acc3);
    }

    // D layout: VGPR v holds row v + 8*(lane>=16), col = lane&15
    const int hf = lane >> 4, l = lane & 15;
    const int colb = cgrp * 64 + l;
    #pragma unroll
    for (int v = 0; v < 8; ++v) {
        int r = rtile * 16 + v + hf * 8;
        if (r < M) {
            float* cr = C + (size_t)r * GAT_EMB + colb;
            cr[0]  = acc0[v];
            cr[16] = acc1[v];
            cr[32] = acc2[v];
            cr[48] = acc3[v];
        }
    }
}

// ------------- per-node attention projections: al = sum_c xp*a -------------
__global__ __launch_bounds__(256)
void al_kernel(const float* __restrict__ xp, const float* __restrict__ a_s,
               const float* __restrict__ a_d, float* __restrict__ als,
               float* __restrict__ ald, int Nn) {
    __shared__ float ss[256];
    __shared__ float sd[256];
    const int n = blockIdx.x;
    const int t = threadIdx.x;
    if (n >= Nn) return;
    float v = xp[(size_t)n * GAT_EMB + t];
    ss[t] = v * a_s[t];
    sd[t] = v * a_d[t];
    for (int s = 32; s > 0; s >>= 1) {
        __syncthreads();
        if ((t & 63) < s) { ss[t] += ss[t + s]; sd[t] += sd[t + s]; }
    }
    __syncthreads();
    if ((t & 63) == 0) {
        als[n * GAT_H + (t >> 6)] = ss[t];
        ald[n * GAT_H + (t >> 6)] = sd[t];
    }
}

// ------------------- per-edge logits (no atomics needed) -------------------
__global__ void logit_kernel(const int* __restrict__ src, const int* __restrict__ dst,
                             const float* __restrict__ als, const float* __restrict__ ald,
                             const float* __restrict__ pe, const float* __restrict__ w_pe,
                             float* __restrict__ logits, int Ee) {
    int e = blockIdx.x * blockDim.x + threadIdx.x;
    if (e >= Ee) return;
    int s = src[e], d = dst[e];
    const float4* ps = (const float4*)(pe + (size_t)s * GAT_PED);
    const float4* pd = (const float4*)(pe + (size_t)d * GAT_PED);
    float acc = 0.f;
    #pragma unroll
    for (int i = 0; i < GAT_PED / 4; ++i) {
        float4 a = ps[i], b = pd[i];
        float dx = a.x - b.x, dy = a.y - b.y, dz = a.z - b.z, dw = a.w - b.w;
        acc += dx * dx + dy * dy + dz * dz + dw * dw;
    }
    float dist = sqrtf(acc + 1e-8f);
    #pragma unroll
    for (int h = 0; h < GAT_H; ++h) {
        float x  = als[s * GAT_H + h] + ald[d * GAT_H + h];
        float lr = (x > 0.f) ? x : NEG_SLOPE * x;
        logits[(size_t)e * GAT_H + h] = lr + dist * w_pe[h];
    }
}

// -------- per-dst-node segment softmax + aggregation (gather, 0 atomics) ---
__global__ __launch_bounds__(256)
void node_softmax_agg(const int* __restrict__ elist, const int* __restrict__ start,
                      const int* __restrict__ src, const float* __restrict__ logits,
                      const float* __restrict__ xp, float* __restrict__ out, int Nn) {
    __shared__ float red[128];
    __shared__ float mh[GAT_H];
    __shared__ float dh[GAT_H];
    __shared__ float wbuf[64 * GAT_H];
    __shared__ int   sbuf[64];

    const int n = blockIdx.x;
    const int t = threadIdx.x;
    if (n >= Nn) return;
    const int s0 = start[n], s1 = start[n + 1];

    // phase 1: per-head max over incoming edges
    if (t < 128) {
        int h = t & 3;
        float lm = -3.4e38f;
        for (int k = s0 + (t >> 2); k < s1; k += 32)
            lm = fmaxf(lm, logits[(size_t)elist[k] * GAT_H + h]);
        red[t] = lm;
    }
    __syncthreads();
    if (t < 64) red[t] = fmaxf(red[t], red[t + 64]);  __syncthreads();
    if (t < 32) red[t] = fmaxf(red[t], red[t + 32]);  __syncthreads();
    if (t < 16) red[t] = fmaxf(red[t], red[t + 16]);  __syncthreads();
    if (t < 8)  red[t] = fmaxf(red[t], red[t + 8]);   __syncthreads();
    if (t < 4)  mh[t]  = fmaxf(red[t], red[t + 4]);   __syncthreads();

    // phase 2: per-head denominator
    if (t < 128) {
        int h = t & 3;
        float m = mh[h], ls = 0.f;
        for (int k = s0 + (t >> 2); k < s1; k += 32)
            ls += __expf(logits[(size_t)elist[k] * GAT_H + h] - m);
        red[t] = ls;
    }
    __syncthreads();
    if (t < 64) red[t] += red[t + 64];  __syncthreads();
    if (t < 32) red[t] += red[t + 32];  __syncthreads();
    if (t < 16) red[t] += red[t + 16];  __syncthreads();
    if (t < 8)  red[t] += red[t + 8];   __syncthreads();
    if (t < 4)  dh[t]  = red[t] + red[t + 4];  __syncthreads();

    // phase 3: chunked weighted aggregation, 1 output channel per thread
    const int h = t >> 6;
    const float invden = 1.f / (dh[h] + 1e-16f);
    float acc = 0.f;
    const int j = t >> 2, h2 = t & 3;
    for (int base = s0; base < s1; base += 64) {
        int cnt = min(64, s1 - base);
        if (j < cnt) {
            int e = elist[base + j];
            wbuf[j * GAT_H + h2] = __expf(logits[(size_t)e * GAT_H + h2] - mh[h2]);
            if (h2 == 0) sbuf[j] = src[e];
        }
        __syncthreads();
        for (int q = 0; q < cnt; ++q)
            acc += wbuf[q * GAT_H + h] * xp[(size_t)sbuf[q] * GAT_EMB + t];
        __syncthreads();
    }
    out[(size_t)n * GAT_EMB + t] = acc * invden;
}

// ------------------ small 32x32 PE linear + relu (VALU) --------------------
__global__ __launch_bounds__(256)
void pe_linear_kernel(const float* __restrict__ in, const float* __restrict__ W,
                      const float* __restrict__ bias, float* __restrict__ out, int Nn) {
    __shared__ float Ws[GAT_PED * GAT_PED];
    __shared__ float bs[GAT_PED];
    int t = threadIdx.x;
    for (int i = t; i < GAT_PED * GAT_PED; i += 256) Ws[i] = W[i];
    if (t < GAT_PED) bs[t] = bias ? bias[t] : 0.f;
    __syncthreads();
    int node = blockIdx.x * 8 + (t >> 5);
    int col  = t & 31;
    if (node >= Nn) return;
    const float* row = in + (size_t)node * GAT_PED;
    float acc = bs[col];
    #pragma unroll
    for (int k = 0; k < GAT_PED; ++k) acc += row[k] * Ws[k * GAT_PED + col];
    out[(size_t)node * GAT_PED + col] = fmaxf(acc, 0.f);
}

// ---------------------------------------------------------------------------
extern "C" void kernel_launch(void* const* d_in, const int* in_sizes, int n_in,
                              void* d_out, int out_size, void* d_ws, size_t ws_size,
                              hipStream_t stream) {
    const int FEAT = 128;
    const int N = in_sizes[0] / FEAT;      // 50000
    const int E = in_sizes[4] / 2;         // 800000

    const float* x_masked = (const float*)d_in[1];
    const float* PE       = (const float*)d_in[2];
    const float* PE_noise = (const float*)d_in[3];
    const int*   eidx     = (const int*)d_in[4];
    const int*   src      = eidx;
    const int*   dst      = eidx + E;
    const float* W_peg = (const float*)d_in[5];
    const float* b_peg = (const float*)d_in[6];
    const float* W0    = (const float*)d_in[7];
    const float* a_s0  = (const float*)d_in[8];
    const float* a_d0  = (const float*)d_in[9];
    const float* wpe0  = (const float*)d_in[10];
    const float* W_u0  = (const float*)d_in[11];
    const float* W1    = (const float*)d_in[12];
    const float* a_s1  = (const float*)d_in[13];
    const float* a_d1  = (const float*)d_in[14];
    const float* wpe1  = (const float*)d_in[15];
    const float* W_u1  = (const float*)d_in[16];

    float* out_h  = (float*)d_out;                // [N, 256]
    float* out_pe = out_h + (size_t)N * GAT_EMB;  // [N, 32]

    // ------------------------- workspace carve-up --------------------------
    char* wp = (char*)d_ws;
    auto alloc = [&](size_t bytes) -> char* {
        char* r = wp;
        wp += (bytes + 255) & ~(size_t)255;
        return r;
    };
    float*  xp     = (float*)alloc((size_t)N * GAT_EMB * 4);
    float*  agg    = (float*)alloc((size_t)N * GAT_EMB * 4);
    float*  als    = (float*)alloc((size_t)N * GAT_H * 4);
    float*  ald    = (float*)alloc((size_t)N * GAT_H * 4);
    float*  pe0    = (float*)alloc((size_t)N * GAT_PED * 4);
    float*  pe1    = (float*)alloc((size_t)N * GAT_PED * 4);
    float*  pen_a  = (float*)alloc((size_t)N * GAT_PED * 4);
    float*  pen_b  = (float*)alloc((size_t)N * GAT_PED * 4);
    float*  logits = (float*)alloc((size_t)E * GAT_H * 4);
    half_t* Aswz   = (half_t*)alloc((size_t)N * GAT_EMB * 2);   // max K=256
    half_t* Bswz   = (half_t*)alloc((size_t)GAT_EMB * GAT_EMB * 2);
    int*    deg    = (int*)alloc((size_t)N * 4);
    int*    startp = (int*)alloc((size_t)(N + 1) * 4);
    int*    cursor = (int*)alloc((size_t)N * 4);
    int*    elist  = (int*)alloc((size_t)E * 4);

    const int peBlocks   = (N + 7) / 8;
    const int gemmBlocks = (N + 31) / 32;
    const int edgeBlocks = (E + 255) / 256;

    // ----------------------- CSR build (once, reused) ----------------------
    fill_i32<<<(N + 255) / 256, 256, 0, stream>>>(deg, 0, N);
    fill_i32<<<(N + 255) / 256, 256, 0, stream>>>(cursor, 0, N);
    deg_kernel<<<edgeBlocks, 256, 0, stream>>>(dst, deg, E);
    scan_kernel<<<1, 1024, 0, stream>>>(deg, startp, N);
    scatter_kernel<<<edgeBlocks, 256, 0, stream>>>(dst, startp, cursor, elist, E);

    // ------------------------ PE encoder + updates -------------------------
    pe_linear_kernel<<<peBlocks, 256, 0, stream>>>(PE,       W_peg, b_peg, pe0,   N);
    pe_linear_kernel<<<peBlocks, 256, 0, stream>>>(PE_noise, W_peg, b_peg, pen_a, N);
    pe_linear_kernel<<<peBlocks, 256, 0, stream>>>(pe0,   W_u0, nullptr, pe1,    N);
    pe_linear_kernel<<<peBlocks, 256, 0, stream>>>(pen_a, W_u0, nullptr, pen_b,  N);
    pe_linear_kernel<<<peBlocks, 256, 0, stream>>>(pen_b, W_u1, nullptr, out_pe, N);

    // ------------------------------ layer 0 --------------------------------
    swizzleA_kernel<<<(N * FEAT + 255) / 256, 256, 0, stream>>>(x_masked, Aswz, N, FEAT, 0);
    swizzleB_kernel<<<(FEAT * GAT_EMB + 255) / 256, 256, 0, stream>>>(W0, Bswz, FEAT);
    gemm_wmma_swz<<<gemmBlocks, 256, 0, stream>>>(Aswz, Bswz, xp, N, FEAT);
    al_kernel<<<N, 256, 0, stream>>>(xp, a_s0, a_d0, als, ald, N);
    logit_kernel<<<edgeBlocks, 256, 0, stream>>>(src, dst, als, ald, pe0, wpe0, logits, E);
    node_softmax_agg<<<N, 256, 0, stream>>>(elist, startp, src, logits, xp, agg, N);

    // ------------------------------ layer 1 --------------------------------
    swizzleA_kernel<<<(N * GAT_EMB + 255) / 256, 256, 0, stream>>>(agg, Aswz, N, GAT_EMB, 1);
    swizzleB_kernel<<<(GAT_EMB * GAT_EMB + 255) / 256, 256, 0, stream>>>(W1, Bswz, GAT_EMB);
    gemm_wmma_swz<<<gemmBlocks, 256, 0, stream>>>(Aswz, Bswz, xp, N, GAT_EMB);
    al_kernel<<<N, 256, 0, stream>>>(xp, a_s1, a_d1, als, ald, N);
    logit_kernel<<<edgeBlocks, 256, 0, stream>>>(src, dst, als, ald, pe1, wpe1, logits, E);
    node_softmax_agg<<<N, 256, 0, stream>>>(elist, startp, src, logits, xp, out_h, N);
}
